// GraphTransformerWOPosScoreModel_23029614641882
// MI455X (gfx1250) — matrix operand order repaired and checked by reading, not compiled, not run on hardware
//
#include <hip/hip_runtime.h>
#include <hip/hip_bf16.h>

// ---------------------------------------------------------------------------
// GraphTransformer forward for MI455X (gfx1250, wave32).
// Dense GEMMs use v_wmma_f32_16x16x32_bf16 (bf16 in, fp32 accumulate).
// Irregular segment ops (edge softmax over dst, GraphNorm) use f32 atomics.
// ---------------------------------------------------------------------------

typedef __attribute__((ext_vector_type(16))) __bf16 v16bf;
typedef __attribute__((ext_vector_type(8)))  __bf16 v8bf;
typedef __attribute__((ext_vector_type(8)))  float  v8f;

#define HID 128          // hidden
#define NHEAD 8
#define CHD 16           // channels per head
#define NLAYER 5

__device__ __forceinline__ float dev_silu(float x) { return x / (1.0f + __expf(-x)); }

// ------------------------- ordered-int float encoding for atomic max -------
__device__ __forceinline__ unsigned enc_ord(float f) {
    unsigned u = __float_as_uint(f);
    unsigned m = (u & 0x80000000u) ? 0xFFFFFFFFu : 0x80000000u;
    return u ^ m;
}
__device__ __forceinline__ float dec_ord(unsigned u) {
    unsigned m = (u & 0x80000000u) ? 0x80000000u : 0xFFFFFFFFu;
    return __uint_as_float(u ^ m);
}
// enc_ord(-inf) == 0x007FFFFF
#define ENC_NEG_INF 0x007FFFFFu

// ------------------------- generic fill ------------------------------------
__global__ void fill_u32_kernel(unsigned* __restrict__ p, unsigned v, long n) {
    long i = (long)blockIdx.x * blockDim.x + threadIdx.x;
    if (i < n) p[i] = v;
}

// ------------------------- weight swizzle (fp32 [K,N] -> WMMA bf16) --------
// Output layout: [ktile][ntile][lane(32)][elem(16)] so each lane's B fragment
// is one contiguous 32B load.  lane: n = ntile*16 + (lane&15),
// elem i: k = ktile*32 + (lane>>4)*16 + i   (k >= K -> 0 pad)
__global__ void swizzle_w_kernel(const float* __restrict__ W, __bf16* __restrict__ out,
                                 int K, int N, int Kpad) {
    int idx = blockIdx.x * blockDim.x + threadIdx.x;
    int total = Kpad * N;                     // (Kpad/32)*(N/16)*512
    if (idx >= total) return;
    int i     = idx & 15;
    int lane  = (idx >> 4) & 31;
    int tile  = idx >> 9;
    int ntiles = N >> 4;
    int ntile = tile % ntiles;
    int ktile = tile / ntiles;
    int n = (ntile << 4) + (lane & 15);
    int k = (ktile << 5) + ((lane >> 4) << 4) + i;
    float w = (k < K) ? W[(size_t)k * N + n] : 0.0f;
    out[idx] = (__bf16)w;
}

// ------------------------- WMMA GEMM: Y[M,N] = act(A[M,K] * Wswz + bias) ---
// A: row-major bf16.  One wave -> one 16x16 tile; block = (32,4) waves.
__global__ void wmma_gemm_kernel(const __bf16* __restrict__ A,
                                 const __bf16* __restrict__ Wsw,
                                 const float* __restrict__ bias,
                                 float* __restrict__ Yf, __bf16* __restrict__ Yb,
                                 int M, int K, int N, int act) {
    int lane  = threadIdx.x;
    int ntile = blockIdx.y * blockDim.y + threadIdx.y;
    int mtile = blockIdx.x;
    int m0 = mtile << 4, n0 = ntile << 4;
    int half = lane >> 4;
    int mr   = lane & 15;
    int nr   = lane & 15;
    int row = m0 + mr; if (row > M - 1) row = M - 1;   // tail-tile clamp
    const __bf16* arow = A + (size_t)row * K + (half << 4);
    const __bf16* bptr = Wsw + ((size_t)ntile * 32 + lane) * 16;
    size_t bstride = (size_t)(N >> 4) * 512;           // elems per ktile
    v8f acc = {};
    int ktiles = K >> 5;
    for (int kt = 0; kt < ktiles; ++kt) {
        v8bf alo = *(const v8bf*)(arow);
        v8bf ahi = *(const v8bf*)(arow + 8);
        v16bf a;
#pragma unroll
        for (int i = 0; i < 8; ++i) { a[i] = alo[i]; a[i + 8] = ahi[i]; }
        v16bf bfr = *(const v16bf*)(bptr);
        acc = __builtin_amdgcn_wmma_f32_16x16x32_bf16(
            false, a, false, bfr, (short)0, acc, false, false);
        arow += 32;
        bptr += bstride;
        __builtin_prefetch(arow, 0, 1);                // global_prefetch_b8
        __builtin_prefetch(bptr, 0, 1);
    }
#pragma unroll
    for (int i = 0; i < 8; ++i) {
        int m = m0 + (half << 3) + i;
        if (m >= M) break;
        float v = acc[i] + (bias ? bias[n0 + nr] : 0.0f);
        if (act) v = dev_silu(v);
        size_t o = (size_t)m * N + n0 + nr;
        if (Yf) Yf[o] = v;
        if (Yb) Yb[o] = (__bf16)v;
    }
}

// ------------------------- time embedding ----------------------------------
// tf[b] = silu( [sin,cos](2*pi*t[b]*gfp) @ te_w + te_b )   grid=B, block=128
__global__ void fourier_kernel(const float* __restrict__ t, const float* __restrict__ gfp,
                               const float* __restrict__ te_w, const float* __restrict__ te_b,
                               float* __restrict__ tf) {
    __shared__ float s[128];
    int b = blockIdx.x, j = threadIdx.x;
    float xp = 6.283185307179586f * t[b] * gfp[j & 63];
    s[j] = (j < 64) ? __sinf(xp) : __cosf(xp);
    __syncthreads();
    float acc = te_b[j];
    for (int i = 0; i < 128; ++i) acc += s[i] * te_w[i * 128 + j];
    tf[b * 128 + j] = dev_silu(acc);
}

// Y[b,j] = A[b,:K] @ W[:,j] + bias[j]      grid=B, block=Nout
__global__ void small_gemm_kernel(const float* __restrict__ A, const float* __restrict__ W,
                                  const float* __restrict__ bias, float* __restrict__ Y,
                                  int K, int Nout) {
    int b = blockIdx.x, j = threadIdx.x;
    if (j >= Nout) return;
    float acc = bias ? bias[j] : 0.0f;
    const float* a = A + (size_t)b * K;
    for (int i = 0; i < K; ++i) acc += a[i] * W[(size_t)i * Nout + j];
    Y[(size_t)b * Nout + j] = acc;
}

// h = coords @ in_w + in_b                 grid=N, block=128
__global__ void node_init_kernel(const float* __restrict__ coords, const float* __restrict__ W,
                                 const float* __restrict__ bias, float* __restrict__ h) {
    int n = blockIdx.x, c = threadIdx.x;
    float x = coords[n * 3], y = coords[n * 3 + 1], z = coords[n * 3 + 2];
    h[(size_t)n * HID + c] = x * W[c] + y * W[HID + c] + z * W[2 * HID + c] + bias[c];
}

// edge attr [E,64] bf16: [gauss(32) | dir(3) | te(16) | pad(13)]  grid=E, block=64
__global__ void edge_attr_kernel(const float* __restrict__ coords,
                                 const int* __restrict__ src, const int* __restrict__ dst,
                                 const float* __restrict__ te, const int* __restrict__ batch,
                                 __bf16* __restrict__ ea, int E) {
    int e = blockIdx.x, c = threadIdx.x;
    if (e >= E) return;
    int s = src[e], d = dst[e];
    float dx = coords[s * 3 + 0] - coords[d * 3 + 0];
    float dy = coords[s * 3 + 1] - coords[d * 3 + 1];
    float dz = coords[s * 3 + 2] - coords[d * 3 + 2];
    float el = sqrtf(dx * dx + dy * dy + dz * dz);
    const float step = 5.0f / 33.0f;  // linspace(0,5,34) step
    float v = 0.0f;
    if (c < 32) {
        float diff = (el - (float)(c + 1) * step) / step;
        v = __expf(-diff * diff) * (1.0f / 1.12f);
    } else if (c < 35) {
        float inv = 1.0f / fmaxf(el, 1e-12f);
        v = (c == 32 ? dx : (c == 33 ? dy : dz)) * inv;
    } else if (c < 51) {
        v = te[batch[s] * 16 + (c - 35)];
    }
    ea[(size_t)e * 64 + c] = (__bf16)v;
}

// x = concat(h, tpn[batch]) -> bf16        grid=N, block=256
__global__ void build_x_kernel(const float* __restrict__ h, const float* __restrict__ tpn,
                               const int* __restrict__ batch, __bf16* __restrict__ xbf) {
    int n = blockIdx.x, c = threadIdx.x;
    float v = (c < HID) ? h[(size_t)n * HID + c]
                        : tpn[(size_t)batch[n] * HID + (c - HID)];
    xbf[(size_t)n * 256 + c] = (__bf16)v;
}

__global__ void f2bf_kernel(const float* __restrict__ a, __bf16* __restrict__ b, long n) {
    long i = (long)blockIdx.x * blockDim.x + threadIdx.x;
    if (i < n) b[i] = (__bf16)a[i];
}

// ------------------------- attention ---------------------------------------
// alpha[e,h] = 0.25 * sum_c q[dst,h,c]*(k[src,h,c]+e[e,h,c]);  atomic-max per dst.
__global__ void attn_alpha_kernel(const float* __restrict__ q, const float* __restrict__ k,
                                  const __bf16* __restrict__ ebuf,
                                  const int* __restrict__ src, const int* __restrict__ dst,
                                  float* __restrict__ alpha, unsigned* __restrict__ amax, int E) {
    long idx = (long)blockIdx.x * blockDim.x + threadIdx.x;
    if (idx >= (long)E * NHEAD) return;
    int ed = (int)(idx >> 3), hd = (int)(idx & 7);
    int s = src[ed], d = dst[ed];
    const float*  qp = q    + (size_t)d  * HID + hd * CHD;
    const float*  kp = k    + (size_t)s  * HID + hd * CHD;
    const __bf16* ep = ebuf + (size_t)ed * HID + hd * CHD;
    float acc = 0.0f;
#pragma unroll
    for (int c = 0; c < CHD; ++c) acc += qp[c] * (kp[c] + (float)ep[c]);
    acc *= 0.25f;  // 1/sqrt(16)
    alpha[idx] = acc;
    atomicMax(&amax[(size_t)d * NHEAD + hd], enc_ord(acc));
}

__global__ void attn_exp_kernel(float* __restrict__ alpha, const unsigned* __restrict__ amax,
                                float* __restrict__ asum, const int* __restrict__ dst, int E) {
    long idx = (long)blockIdx.x * blockDim.x + threadIdx.x;
    if (idx >= (long)E * NHEAD) return;
    int ed = (int)(idx >> 3), hd = (int)(idx & 7);
    int d = dst[ed];
    float a = __expf(alpha[idx] - dec_ord(amax[(size_t)d * NHEAD + hd]));
    alpha[idx] = a;
    atomicAdd(&asum[(size_t)d * NHEAD + hd], a);
}

// agg[dst,c] += (v[src,c]+e[e,c]) * alpha/(asum+eps)   grid=E, block=128
__global__ void attn_agg_kernel(const float* __restrict__ v, const __bf16* __restrict__ ebuf,
                                const float* __restrict__ alpha, const float* __restrict__ asum,
                                const int* __restrict__ src, const int* __restrict__ dst,
                                float* __restrict__ agg, int E) {
    int ed = blockIdx.x, c = threadIdx.x;
    if (ed >= E) return;
    int s = src[ed], d = dst[ed], hd = c >> 4;
    float a = alpha[(size_t)ed * NHEAD + hd] / (asum[(size_t)d * NHEAD + hd] + 1e-16f);
    float m = (v[(size_t)s * HID + c] + (float)ebuf[(size_t)ed * HID + c]) * a;
    atomicAdd(&agg[(size_t)d * HID + c], m);
}

// ------------------------- beta gate + GraphNorm ---------------------------
__global__ void beta_hc_kernel(const float* __restrict__ agg, const float* __restrict__ xr,
                               const float* __restrict__ bw, const int* __restrict__ batch,
                               float* __restrict__ hc, float* __restrict__ gsum) {
    __shared__ float red[HID];
    int n = blockIdx.x, c = threadIdx.x;
    float o = agg[(size_t)n * HID + c];
    float x = xr[(size_t)n * HID + c];
    red[c] = o * bw[c] + x * bw[HID + c] + (o - x) * bw[2 * HID + c];
    __syncthreads();
    for (int st = HID / 2; st > 0; st >>= 1) {
        if (c < st) red[c] += red[c + st];
        __syncthreads();
    }
    float beta = 1.0f / (1.0f + __expf(-red[0]));
    float h = beta * x + (1.0f - beta) * o;
    hc[(size_t)n * HID + c] = h;
    atomicAdd(&gsum[(size_t)batch[n] * HID + c], h);
}

__global__ void norm_var_kernel(float* __restrict__ hc, const float* __restrict__ gsum,
                                const float* __restrict__ gn_ms, const int* __restrict__ batch,
                                float* __restrict__ gvar, float invCnt) {
    int n = blockIdx.x, c = threadIdx.x;
    int b = batch[n];
    float mean = gsum[(size_t)b * HID + c] * invCnt;
    float oc = hc[(size_t)n * HID + c] - mean * gn_ms[c];
    hc[(size_t)n * HID + c] = oc;
    atomicAdd(&gvar[(size_t)b * HID + c], oc * oc);
}

__global__ void norm_fin_kernel(const float* __restrict__ hc, const float* __restrict__ gvar,
                                const float* __restrict__ gn_w, const float* __restrict__ gn_b,
                                const float* __restrict__ hres, const int* __restrict__ batch,
                                float* __restrict__ hout, float invCnt) {
    int n = blockIdx.x, c = threadIdx.x;
    int b = batch[n];
    float var = gvar[(size_t)b * HID + c] * invCnt;
    float hn = gn_w[c] * hc[(size_t)n * HID + c] * rsqrtf(var + 1e-5f) + gn_b[c];
    hout[(size_t)n * HID + c] = dev_silu(hn + hres[(size_t)n * HID + c]);
}

// ------------------------- output head 128 -> 3 ----------------------------
__global__ void out_proj_kernel(const float* __restrict__ h1, const float* __restrict__ W,
                                const float* __restrict__ bias, float* __restrict__ out, int N) {
    int idx = blockIdx.x * blockDim.x + threadIdx.x;
    if (idx >= N * 3) return;
    int n = idx / 3, oc = idx % 3;
    float acc = bias[oc];
    const float* hp = h1 + (size_t)n * HID;
    for (int i = 0; i < HID; ++i) acc += hp[i] * W[i * 3 + oc];
    out[idx] = acc;
}

// ===========================================================================
extern "C" void kernel_launch(void* const* d_in, const int* in_sizes, int n_in,
                              void* d_out, int out_size, void* d_ws, size_t ws_size,
                              hipStream_t stream) {
    (void)n_in; (void)out_size; (void)ws_size;
    // ---- inputs in setup_inputs() insertion order (90 leaves) ----
    const float* coords = (const float*)d_in[0];
    const float* t_in   = (const float*)d_in[1];
    const float* te_w  = (const float*)d_in[2];
    const float* te_b  = (const float*)d_in[3];
    const float* tep_w = (const float*)d_in[4];
    const float* tep_b = (const float*)d_in[5];
    const float* in_w  = (const float*)d_in[6];
    const float* in_b  = (const float*)d_in[7];
    const float* o1_w  = (const float*)d_in[8];
    const float* o1_b  = (const float*)d_in[9];
    const float* o2_w  = (const float*)d_in[10];
    const float* o2_b  = (const float*)d_in[11];
    const float *tm_w[NLAYER], *tm_b[NLAYER], *q_w[NLAYER], *q_b[NLAYER],
                *k_w[NLAYER], *k_b[NLAYER], *v_w[NLAYER], *v_b[NLAYER],
                *e_w[NLAYER], *s_w[NLAYER], *s_b[NLAYER], *b_w[NLAYER],
                *gn_w[NLAYER], *gn_b[NLAYER], *gn_ms[NLAYER];
    for (int l = 0; l < NLAYER; ++l) {
        int base = 12 + 15 * l;
        tm_w[l]  = (const float*)d_in[base + 0];  tm_b[l]  = (const float*)d_in[base + 1];
        q_w[l]   = (const float*)d_in[base + 2];  q_b[l]   = (const float*)d_in[base + 3];
        k_w[l]   = (const float*)d_in[base + 4];  k_b[l]   = (const float*)d_in[base + 5];
        v_w[l]   = (const float*)d_in[base + 6];  v_b[l]   = (const float*)d_in[base + 7];
        e_w[l]   = (const float*)d_in[base + 8];
        s_w[l]   = (const float*)d_in[base + 9];  s_b[l]   = (const float*)d_in[base + 10];
        b_w[l]   = (const float*)d_in[base + 11];
        gn_w[l]  = (const float*)d_in[base + 12]; gn_b[l]  = (const float*)d_in[base + 13];
        gn_ms[l] = (const float*)d_in[base + 14];
    }
    const float* gfp_w = (const float*)d_in[87];
    const int*   batch = (const int*)d_in[88];
    const int*   eidx  = (const int*)d_in[89];

    const int N = in_sizes[0] / 3;
    const int B = in_sizes[1];
    const int E = in_sizes[89] / 2;
    const int* esrc = eidx;
    const int* edst = eidx + E;
    const float invCnt = (float)B / (float)N;   // 1 / nodes-per-graph

    // ---- workspace carve-out ----
    char* wsb = (char*)d_ws;
    size_t off = 0;
    auto alloc = [&](size_t bytes) -> char* {
        char* p = wsb + off;
        off += (bytes + 255) & ~(size_t)255;
        return p;
    };
    float*    tf    = (float*)alloc((size_t)B * HID * 4);
    float*    tep   = (float*)alloc((size_t)B * 16 * 4);
    float*    tpn   = (float*)alloc((size_t)NLAYER * B * HID * 4);
    float*    hA    = (float*)alloc((size_t)N * HID * 4);
    float*    hB    = (float*)alloc((size_t)N * HID * 4);
    __bf16*   xbf   = (__bf16*)alloc((size_t)N * 256 * 2);
    __bf16*   hbf   = (__bf16*)alloc((size_t)N * HID * 2);
    float*    qbuf  = (float*)alloc((size_t)N * HID * 4);
    float*    kbuf  = (float*)alloc((size_t)N * HID * 4);
    float*    vbuf  = (float*)alloc((size_t)N * HID * 4);
    float*    xrbuf = (float*)alloc((size_t)N * HID * 4);
    __bf16*   eattr = (__bf16*)alloc((size_t)E * 64 * 2);
    __bf16*   ebuf  = (__bf16*)alloc((size_t)E * HID * 2);
    float*    alpha = (float*)alloc((size_t)E * NHEAD * 4);
    unsigned* amax  = (unsigned*)alloc((size_t)N * NHEAD * 4);
    float*    asum  = (float*)alloc((size_t)N * NHEAD * 4);
    float*    agg   = (float*)alloc((size_t)N * HID * 4);
    float*    hc    = (float*)alloc((size_t)N * HID * 4);
    float*    gsum  = (float*)alloc((size_t)B * HID * 4);
    float*    gvar  = (float*)alloc((size_t)B * HID * 4);
    float*    h1    = (float*)alloc((size_t)N * HID * 4);
    __bf16 *wq[NLAYER], *wk[NLAYER], *wv[NLAYER], *wsk[NLAYER], *we[NLAYER];
    for (int l = 0; l < NLAYER; ++l) {
        wq[l]  = (__bf16*)alloc((size_t)256 * HID * 2);
        wk[l]  = (__bf16*)alloc((size_t)256 * HID * 2);
        wv[l]  = (__bf16*)alloc((size_t)256 * HID * 2);
        wsk[l] = (__bf16*)alloc((size_t)256 * HID * 2);
        we[l]  = (__bf16*)alloc((size_t)64 * HID * 2);
    }
    __bf16* wo1 = (__bf16*)alloc((size_t)HID * HID * 2);

    auto fillu = [&](void* p, unsigned v, long n) {
        fill_u32_kernel<<<(unsigned)((n + 255) / 256), 256, 0, stream>>>((unsigned*)p, v, n);
    };
    auto swz = [&](const float* W, __bf16* O, int K, int Ncol, int Kpad) {
        int total = Kpad * Ncol;
        swizzle_w_kernel<<<(total + 255) / 256, 256, 0, stream>>>(W, O, K, Ncol, Kpad);
    };

    // ---- weight conversion + swizzle (once per call) ----
    for (int l = 0; l < NLAYER; ++l) {
        swz(q_w[l], wq[l], 256, HID, 256);
        swz(k_w[l], wk[l], 256, HID, 256);
        swz(v_w[l], wv[l], 256, HID, 256);
        swz(s_w[l], wsk[l], 256, HID, 256);
        swz(e_w[l], we[l], 51, HID, 64);     // pad K 51 -> 64
    }
    swz(o1_w, wo1, HID, HID, HID);

    // ---- time embeddings ----
    fourier_kernel<<<B, 128, 0, stream>>>(t_in, gfp_w, te_w, te_b, tf);
    small_gemm_kernel<<<B, 16, 0, stream>>>(tf, tep_w, tep_b, tep, HID, 16);
    for (int l = 0; l < NLAYER; ++l)
        small_gemm_kernel<<<B, HID, 0, stream>>>(tf, tm_w[l], tm_b[l],
                                                 tpn + (size_t)l * B * HID, HID, HID);

    // ---- node init + edge features ----
    node_init_kernel<<<N, HID, 0, stream>>>(coords, in_w, in_b, hA);
    edge_attr_kernel<<<E, 64, 0, stream>>>(coords, esrc, edst, tep, batch, eattr, E);

    // ---- transformer layers ----
    dim3 blk(32, 4);                       // 4 waves, each owns one 16x16 tile
    dim3 gnode((N + 15) / 16, HID / 64);   // N tiles x 2
    dim3 gedge((E + 15) / 16, HID / 64);
    long EH = (long)E * NHEAD;
    float* hcur = hA;
    float* hnxt = hB;
    for (int l = 0; l < NLAYER; ++l) {
        build_x_kernel<<<N, 256, 0, stream>>>(hcur, tpn + (size_t)l * B * HID, batch, xbf);
        wmma_gemm_kernel<<<gnode, blk, 0, stream>>>(xbf, wq[l],  q_b[l], qbuf,  nullptr, N, 256, HID, 0);
        wmma_gemm_kernel<<<gnode, blk, 0, stream>>>(xbf, wk[l],  k_b[l], kbuf,  nullptr, N, 256, HID, 0);
        wmma_gemm_kernel<<<gnode, blk, 0, stream>>>(xbf, wv[l],  v_b[l], vbuf,  nullptr, N, 256, HID, 0);
        wmma_gemm_kernel<<<gnode, blk, 0, stream>>>(xbf, wsk[l], s_b[l], xrbuf, nullptr, N, 256, HID, 0);
        wmma_gemm_kernel<<<gedge, blk, 0, stream>>>(eattr, we[l], nullptr, nullptr, ebuf, E, 64, HID, 0);

        fillu(amax, ENC_NEG_INF, (long)N * NHEAD);
        fillu(asum, 0u, (long)N * NHEAD);
        fillu(agg,  0u, (long)N * HID);
        fillu(gsum, 0u, (long)B * HID);
        fillu(gvar, 0u, (long)B * HID);

        attn_alpha_kernel<<<(unsigned)((EH + 255) / 256), 256, 0, stream>>>(
            qbuf, kbuf, ebuf, esrc, edst, alpha, amax, E);
        attn_exp_kernel<<<(unsigned)((EH + 255) / 256), 256, 0, stream>>>(
            alpha, amax, asum, edst, E);
        attn_agg_kernel<<<E, HID, 0, stream>>>(vbuf, ebuf, alpha, asum, esrc, edst, agg, E);

        beta_hc_kernel<<<N, HID, 0, stream>>>(agg, xrbuf, b_w[l], batch, hc, gsum);
        norm_var_kernel<<<N, HID, 0, stream>>>(hc, gsum, gn_ms[l], batch, gvar, invCnt);
        norm_fin_kernel<<<N, HID, 0, stream>>>(hc, gvar, gn_w[l], gn_b[l], hcur, batch,
                                               hnxt, invCnt);
        float* tmp = hcur; hcur = hnxt; hnxt = tmp;
    }

    // ---- output head ----
    f2bf_kernel<<<(unsigned)(((long)N * HID + 255) / 256), 256, 0, stream>>>(
        hcur, hbf, (long)N * HID);
    wmma_gemm_kernel<<<gnode, blk, 0, stream>>>(hbf, wo1, o1_b, h1, nullptr, N, HID, HID, 1);
    out_proj_kernel<<<(N * 3 + 127) / 128, 128, 0, stream>>>(h1, o2_w, o2_b, (float*)d_out, N);
}
